// SSIMLoss_69406671504052
// MI455X (gfx1250) — matrix operand order repaired and checked by reading, not compiled, not run on hardware
//
#include <hip/hip_runtime.h>

typedef __attribute__((ext_vector_type(16))) _Float16 v16h;
typedef __attribute__((ext_vector_type(2)))  _Float16 v2h;
typedef __attribute__((ext_vector_type(2)))  __fp16   v2fp;
typedef __attribute__((ext_vector_type(8)))  float    v8f;

#define KS   11
#define RAD  5
#define IMG  512
#define NIMG 64
#define STRIP 32                 // output rows per wave (2 tiles of 16)
#define A_TILES 3                // 48 halo rows cover 32+10 needed
#define HROWS 42                 // stored H rows per map (tail rows never read)
#define ROWSTRIPS (IMG / STRIP)  // 16
#define COLSTRIPS (IMG / 16)     // 32
#define WPB 8                    // waves per block

__device__ __forceinline__ v8f wmma16(v16h a, v16h b, v8f c) {
  // D(16x16 f32) = A(16x32 f16) x B(32x16 f16) + C
  return __builtin_amdgcn_wmma_f32_16x16x32_f16(false, a, false, b, (short)0, c,
                                                false, false);
}

// v_cvt_pk_rtz_f16_f32: two f32 -> packed f16 pair (bit-identical type pun)
__device__ __forceinline__ v2h cvt_pk(float a, float b) {
  union { v2fp f; v2h h; } u;
  u.f = __builtin_amdgcn_cvt_pkrtz(a, b);
  return u.h;
}

// Gaussian tap weight w[d], d in [0,11), computed arithmetically (no dynamic
// table indexing -> no cndmask select chains).
__device__ __forceinline__ float gweight(int d, float nrm) {
  const float t = (float)(d - RAD);
  const float v = nrm * expf(-(t * t) * (1.0f / 4.5f));
  return (d >= 0 && d < KS) ? v : 0.0f;
}

// ---- 4 aligned float4 loads -> one 16x32 f16 A-tile row (per-lane K map):
//   hi=0: j<8 -> K=j      ; j>=8 -> K=j+8    (K 0..7, 16..23)
//   hi=1: j<8 -> K=j+8    ; j>=8 -> K=j+16   (K 8..15, 24..31)
__device__ __forceinline__ v16h load_a_fast(const float* __restrict__ rowp, int hi) {
  const int k0 = hi ? 8 : 0;
  const int k1 = hi ? 24 : 16;
  float4 p0 = *(const float4*)(rowp + k0);
  float4 p1 = *(const float4*)(rowp + k0 + 4);
  float4 p2 = *(const float4*)(rowp + k1);
  float4 p3 = *(const float4*)(rowp + k1 + 4);
  union { v16h v; v2h p[8]; } u;
  u.p[0] = cvt_pk(p0.x, p0.y);
  u.p[1] = cvt_pk(p0.z, p0.w);
  u.p[2] = cvt_pk(p1.x, p1.y);
  u.p[3] = cvt_pk(p1.z, p1.w);
  u.p[4] = cvt_pk(p2.x, p2.y);
  u.p[5] = cvt_pk(p2.z, p2.w);
  u.p[6] = cvt_pk(p3.x, p3.y);
  u.p[7] = cvt_pk(p3.z, p3.w);
  return u.v;
}

__global__ __launch_bounds__(256) void ssim_strip_kernel(
    const float* __restrict__ x, const float* __restrict__ y,
    double* __restrict__ accum) {
  __shared__ _Float16 Hlds[WPB][5][HROWS][16];  // 53760 B
  __shared__ _Float16 WB[32][16];               // 1 KB horizontal B operand
  __shared__ _Float16 WA[3][16][32];            // 3 KB vertical A operand (3 variants)

  const int tid  = threadIdx.x;
  const int lane = tid & 31;
  const int wvid = tid >> 5;
  const int lm   = lane & 15;
  const int hi   = lane >> 4;

  // ---- block geometry: col0 is block-uniform so WB can be shared ----
  const int b    = blockIdx.x;
  const int img  = b / (COLSTRIPS * 2);
  const int rem  = b % (COLSTRIPS * 2);
  const int col0 = (rem >> 1) * 16;
  const int rg   = rem & 1;
  const int row0 = (rg * WPB + wvid) * STRIP;   // per-wave row strip

  // Clamp the 32-column window into the image; fold the shift into the band.
  // Out-of-image columns == taps that fall off the K=32 window => free zero pad.
  const int c0s     = min(max(col0 - 8, 0), IMG - 32);
  const int bandoff = col0 - RAD - c0s;         // -5 (left) / 3 / 11 (right)

  // ---- 1/sum(g) (constant-folds) ----
  float nrm;
  {
    float s = 0.f;
#pragma unroll
    for (int t = 0; t < KS; ++t) {
      const float d = (float)t - (float)RAD;
      s += expf(-(d * d) * (1.0f / 4.5f));
    }
    nrm = 1.0f / s;
  }

  // ---- fill weight-operand tables (8 entries/thread, once per block) ----
  {
    // WB[k][n] = w[k - n - bandoff]  (256 pairs)
    const int kk = tid >> 3;
    const int n0 = (tid & 7) * 2;
    WB[kk][n0]     = (_Float16)gweight(kk - n0 - bandoff, nrm);
    WB[kk][n0 + 1] = (_Float16)gweight(kk - n0 - 1 - bandoff, nrm);
    // WA[v][m][k] = w[k - m]; v1: zero k<5 (top strip, window 0);
    //                         v2: zero k>20 (bottom strip, window 1)
    const int m  = tid >> 4;
    const int k0 = (tid & 15) * 2;
#pragma unroll
    for (int v = 0; v < 3; ++v) {
#pragma unroll
      for (int q = 0; q < 2; ++q) {
        const int k = k0 + q;
        float val = gweight(k - m, nrm);
        if (v == 1 && k < RAD) val = 0.f;
        if (v == 2 && k > 20)  val = 0.f;
        WA[v][m][k] = (_Float16)val;
      }
    }
  }
  __syncthreads();

  // ---- horizontal B operand: lane = n column, 16 K values in table order
  union { v16h v; _Float16 h[16]; } uwb;
#pragma unroll
  for (int j = 0; j < 16; ++j) uwb.h[j] = WB[j < 8 ? 8 * hi + j : 8 + 8 * hi + j][lm];
  const v16h wBv = uwb.v;

  // ---- branchless halo loads: rows edge-replicated (killed by WA variants) --
  const float* xi = x + (size_t)img * IMG * IMG;
  const float* yi = y + (size_t)img * IMG * IMG;
  v16h ax[A_TILES], ay[A_TILES];
#pragma unroll
  for (int s = 0; s < A_TILES; ++s) {
    int r = row0 - RAD + 16 * s + lm;
    r = min(max(r, 0), IMG - 1);
    const size_t off = (size_t)r * IMG + c0s;
    ax[s] = load_a_fast(xi + off, hi);
    ay[s] = load_a_fast(yi + off, hi);
  }

  // ---- phase 1: horizontal pass for all 5 maps -> LDS ----
#pragma unroll
  for (int map = 0; map < 5; ++map) {
#pragma unroll
    for (int s = 0; s < A_TILES; ++s) {
      v16h a;
      if (map == 0)      a = ax[s];
      else if (map == 1) a = ay[s];
      else if (map == 2) a = ax[s] * ax[s];
      else if (map == 3) a = ay[s] * ay[s];
      else               a = ax[s] * ay[s];

      const v8f z = {};
      v8f h = wmma16(a, wBv, z);
      // D layout: value r -> (m = r + 8*hi, n = lm); H row = 16s + m
#pragma unroll
      for (int q = 0; q < 4; ++q) {
        const v2h p = cvt_pk(h[2 * q], h[2 * q + 1]);
        const int row = 16 * s + 2 * q + 8 * hi;
        const bool ok = (s < 2) || (2 * q + 8 * hi) < (HROWS - 32);  // row+1 < 42
        if (ok) {
          Hlds[wvid][map][row][lm]     = p.x;
          Hlds[wvid][map][row + 1][lm] = p.y;
        }
      }
    }
  }

  // ---- phase 2: vertical pass + SSIM per 16x16 tile ----
  const float C1 = 1e-4f, C2 = 9e-4f;
  float acc = 0.f;
  const int k0 = hi ? 8 : 0;
  const int k1 = hi ? 24 : 16;
#pragma unroll
  for (int t = 0; t < STRIP / 16; ++t) {
    int vidx = 0;
    if (t == 0 && row0 == 0)           vidx = 1;
    if (t == 1 && row0 == IMG - STRIP) vidx = 2;

    union { v16h v; _Float16 h[16]; } uwa;
#pragma unroll
    for (int j = 0; j < 8; ++j) {
      uwa.h[j]     = WA[vidx][lm][k0 + j];
      uwa.h[8 + j] = WA[vidx][lm][k1 + j];
    }
    const v16h wAv = uwa.v;

    int rrow = 16 * t;                    // H row of K=0; element j -> K in table order, column lm

    v8f conv[5];
#pragma unroll
    for (int map = 0; map < 5; ++map) {
      union { v16h v; _Float16 h[16]; } ub;
#pragma unroll
      for (int j = 0; j < 16; ++j) ub.h[j] = Hlds[wvid][map][min(rrow + (j < 8 ? k0 + j : k1 + j - 8), HROWS - 1)][lm];  // tail rows are zero-weighted
      const v8f z = {};
      conv[map] = wmma16(wAv, ub.v, z);
    }
#pragma unroll
    for (int r = 0; r < 8; ++r) {
      const float mx  = conv[0][r], my = conv[1][r];
      const float sxx = conv[2][r] - mx * mx;
      const float syy = conv[3][r] - my * my;
      const float sxy = conv[4][r] - mx * my;
      const float num = (2.f * mx * my + C1) * (2.f * sxy + C2);
      const float den = (mx * mx + my * my + C1) * (sxx + syy + C2) + 1e-12f;
      acc += num / den;
    }
  }

  // ---- wave32 reduction + one f64 atomic per wave ----
#pragma unroll
  for (int off = 16; off > 0; off >>= 1) acc += __shfl_down(acc, off, 32);
  if (lane == 0) atomicAdd(accum, (double)acc);
}

__global__ void ssim_finalize(const double* __restrict__ accum,
                              float* __restrict__ out) {
  const double n = (double)((size_t)NIMG * IMG * IMG);
  out[0] = (float)(1.0 - accum[0] / n);
}

extern "C" void kernel_launch(void* const* d_in, const int* in_sizes, int n_in,
                              void* d_out, int out_size, void* d_ws, size_t ws_size,
                              hipStream_t stream) {
  (void)in_sizes; (void)n_in; (void)out_size; (void)ws_size;
  const float* x = (const float*)d_in[0];
  const float* y = (const float*)d_in[1];
  double* accum = (double*)d_ws;

  (void)hipMemsetAsync(accum, 0, sizeof(double), stream);

  const int blocks = NIMG * COLSTRIPS * 2;   // 4096 blocks x 256 threads
  ssim_strip_kernel<<<blocks, 256, 0, stream>>>(x, y, accum);
  ssim_finalize<<<1, 1, 0, stream>>>(accum, (float*)d_out);
}
